// SelfAttention_62388694942143
// MI455X (gfx1250) — matrix operand order repaired and checked by reading, not compiled, not run on hardware
//
#include <hip/hip_runtime.h>
#include <hip/hip_bf16.h>

typedef __bf16 bf16_t;
typedef __attribute__((ext_vector_type(16))) __bf16 v16bf;
typedef __attribute__((ext_vector_type(8)))  __bf16 bfx8;
typedef __attribute__((ext_vector_type(8)))  float  v8f;
typedef __attribute__((ext_vector_type(4)))  float  f32x4;

static constexpr int Bsz = 2, Lseq = 2048, DIM = 2048, NH = 16, HD = 128;

__device__ __forceinline__ v8f wmma_bf16(v16bf a, v16bf b, v8f c) {
  // D = A(16x32 bf16) * B(32x16 bf16) + C(16x16 f32)
  return __builtin_amdgcn_wmma_f32_16x16x32_bf16(false, a, false, b, (short)0, c,
                                                 false, false);
}

// Async global->LDS 16B copy (per lane). VDST = LDS byte address (low 32 bits
// of the generic shared pointer), VADDR = 64-bit global address. ASYNCcnt.
__device__ __forceinline__ void async_cp16(void* lds_dst, const void* gsrc) {
  asm volatile("global_load_async_to_lds_b128 %0, %1, off"
               :: "v"((unsigned int)(uintptr_t)lds_dst), "v"(gsrc)
               : "memory");
}
__device__ __forceinline__ void wait_async0() {
  asm volatile("s_wait_asynccnt 0x0" ::: "memory");
}
__device__ __forceinline__ void wait_async8() {   // allow 8 newest in flight
  asm volatile("s_wait_asynccnt 0x8" ::: "memory");
}

// A fragment (16x32, bf16): lane 0-15 -> row=lane, K-half 0; lane 16-31 -> same
// rows, K-half 8. v0..3 = K[kh..kh+7], v4..7 = K[16+kh..16+kh+7].
__device__ __forceinline__ v16bf frag_a(const bf16_t* base, int stride, int lane) {
  const int row = lane & 15;
  const int kh  = (lane >> 4) << 3;           // 0 or 8
  const bf16_t* p = base + row * stride;
  bfx8 lo = *(const bfx8*)(p + kh);
  bfx8 hi = *(const bfx8*)(p + 16 + kh);
  v16bf r;
#pragma unroll
  for (int i = 0; i < 8; ++i) { r[i] = lo[i]; r[i + 8] = hi[i]; }
  return r;
}

// B fragment (32x16, bf16), LDS laid out [col][k]: lane 0-15 -> col=lane,
// K-half 0; lane 16-31 -> col=lane-16, K-half 8. v0..3 = K[kh..kh+7], v4..7 = K[16+kh..16+kh+7].
__device__ __forceinline__ v16bf frag_b(const bf16_t* base, int stride, int lane) {
  const int col = lane & 15;
  const int kh  = (lane >> 4) << 3;           // 0 or 8
  const bf16_t* p = base + col * stride;
  bfx8 lo = *(const bfx8*)(p + kh);
  bfx8 hi = *(const bfx8*)(p + 16 + kh);
  v16bf r;
#pragma unroll
  for (int i = 0; i < 8; ++i) { r[i] = lo[i]; r[i + 8] = hi[i]; }
  return r;
}

// ---------------------------------------------------------------------------
// GEMM: C[M,N] = A[M,K] * W[N,K]^T (+bias). A fp32 or bf16, W fp32, C fp32.
// WG tile 128x128, BK=64 (2 WMMA K-steps per stage), 8 waves (4Mx2N), wave
// tile 32x64 -> 16 WMMAs per stage. Global fetch for stage k+1 (and the
// fp32->bf16 convert) is register-pipelined across stage k's WMMAs.
// ---------------------------------------------------------------------------
template <bool A_BF16, bool BIAS>
__global__ __launch_bounds__(256, 2) void gemm_wmma(
    const void* __restrict__ Ap, const float* __restrict__ Bw,
    const float* __restrict__ bias, float* __restrict__ C,
    int M, int N, int K) {
  constexpr int LDT = 72;                     // 64 + 8 pad, 144B rows (16B-aligned)
  __shared__ bf16_t sA[128 * LDT];
  __shared__ bf16_t sB[128 * LDT];

  const int tid = threadIdx.x, lane = tid & 31, wave = tid >> 5;
  const int wm = wave & 3, wn = wave >> 2;
  const int bm = blockIdx.y * 128, bn = blockIdx.x * 128;

  v8f acc[2][4];
#pragma unroll
  for (int i = 0; i < 2; ++i)
#pragma unroll
    for (int j = 0; j < 4; ++j)
#pragma unroll
      for (int r = 0; r < 8; ++r) acc[i][j][r] = 0.0f;

  const int lr = tid >> 3;                    // 0..31
  const int lc = (tid & 7) * 8;               // 0..56

  bfx8 ra[4], rb[4];                          // staging registers

  auto fetch = [&](int k0) {
#pragma unroll
    for (int p = 0; p < 4; ++p) {
      const int row = lr + p * 32;
      if constexpr (A_BF16) {
        ra[p] = *(const bfx8*)((const bf16_t*)Ap + (size_t)(bm + row) * K + k0 + lc);
      } else {
        const float* ga = (const float*)Ap + (size_t)(bm + row) * K + k0 + lc;
        f32x4 t0 = *(const f32x4*)(ga);
        f32x4 t1 = *(const f32x4*)(ga + 4);
#pragma unroll
        for (int i = 0; i < 4; ++i) { ra[p][i] = (bf16_t)t0[i]; ra[p][i + 4] = (bf16_t)t1[i]; }
      }
      const float* gb = Bw + (size_t)(bn + row) * K + k0 + lc;
      f32x4 b0 = *(const f32x4*)(gb);
      f32x4 b1 = *(const f32x4*)(gb + 4);
#pragma unroll
      for (int i = 0; i < 4; ++i) { rb[p][i] = (bf16_t)b0[i]; rb[p][i + 4] = (bf16_t)b1[i]; }
    }
  };
  auto commit = [&]() {
#pragma unroll
    for (int p = 0; p < 4; ++p) {
      const int row = lr + p * 32;
      *(bfx8*)(&sA[row * LDT + lc]) = ra[p];
      *(bfx8*)(&sB[row * LDT + lc]) = rb[p];
    }
  };

  fetch(0);
  commit();
  for (int k0 = 0; k0 < K; k0 += 64) {
    __syncthreads();                          // stage k0 visible in LDS
    const bool more = (k0 + 64) < K;
    if (more) fetch(k0 + 64);                 // overlap next fetch with WMMAs

    const bf16_t* Abase = sA + (wm * 32) * LDT;
    const bf16_t* Bbase = sB + (wn * 64) * LDT;
#pragma unroll
    for (int ks = 0; ks < 2; ++ks) {
      const v16bf af0 = frag_a(Abase + ks * 32, LDT, lane);
      const v16bf af1 = frag_a(Abase + 16 * LDT + ks * 32, LDT, lane);
#pragma unroll
      for (int nt = 0; nt < 4; ++nt) {
        const v16bf bf = frag_b(Bbase + nt * 16 * LDT + ks * 32, LDT, lane);
        acc[0][nt] = wmma_bf16(af0, bf, acc[0][nt]);
        acc[1][nt] = wmma_bf16(af1, bf, acc[1][nt]);
      }
    }
    __syncthreads();                          // all reads of stage k0 done
    if (more) commit();                       // publish stage k0+64
  }

  // C/D layout: VGPR r -> row r (lanes 0-15) / row r+8 (lanes 16-31), col=lane&15.
  const int rhi = (lane >> 4) << 3;
#pragma unroll
  for (int mt = 0; mt < 2; ++mt) {
#pragma unroll
    for (int nt = 0; nt < 4; ++nt) {
      const int row0 = bm + wm * 32 + mt * 16 + rhi;
      const int col  = bn + wn * 64 + nt * 16 + (lane & 15);
      const float bv = BIAS ? bias[col] : 0.0f;
#pragma unroll
      for (int r = 0; r < 8; ++r)
        C[(size_t)(row0 + r) * N + col] = acc[mt][nt][r] + bv;
    }
  }
}

// ---------------------------------------------------------------------------
// Per-head RMSNorm of Q/K + layout change. One wave per (b,l,{q,k,v},h) row.
// Writes qn/kn as bf16 [B,H,L,D]; V transposed to [B,H,D,L] for PV fragments.
// ---------------------------------------------------------------------------
__global__ __launch_bounds__(256) void rms_reshape(
    const float* __restrict__ qkv, const float* __restrict__ qw,
    const float* __restrict__ kw, bf16_t* __restrict__ qn,
    bf16_t* __restrict__ kn, bf16_t* __restrict__ vT) {
  const int lane = threadIdx.x & 31, wave = threadIdx.x >> 5;
  const size_t rowid = (size_t)blockIdx.x * 8 + wave;  // over B*L*3*H
  const int h = (int)(rowid % NH);
  size_t t = rowid / NH;
  const int s = (int)(t % 3); t /= 3;
  const int l = (int)(t % Lseq);
  const int b = (int)(t / Lseq);

  f32x4 v = *(const f32x4*)(qkv + rowid * HD + lane * 4);
  float ss = v[0] * v[0] + v[1] * v[1] + v[2] * v[2] + v[3] * v[3];
#pragma unroll
  for (int m = 1; m < 32; m <<= 1) ss += __shfl_xor(ss, m, 32);
  const float inv = rsqrtf(ss * (1.0f / HD) + 1e-6f);

  const size_t bh = (size_t)b * NH + h;
  const int d0 = lane * 4;
  if (s == 0) {
#pragma unroll
    for (int i = 0; i < 4; ++i)
      qn[(bh * Lseq + l) * HD + d0 + i] = (bf16_t)(v[i] * inv * qw[d0 + i]);
  } else if (s == 1) {
#pragma unroll
    for (int i = 0; i < 4; ++i)
      kn[(bh * Lseq + l) * HD + d0 + i] = (bf16_t)(v[i] * inv * kw[d0 + i]);
  } else {
#pragma unroll
    for (int i = 0; i < 4; ++i)
      vT[(bh * HD + d0 + i) * Lseq + l] = (bf16_t)v[i];
  }
}

// ---------------------------------------------------------------------------
// Flash attention: WG = 128 q rows (16 per wave), stream 64-key K/V tiles.
// K/V tiles stream through double-buffered LDS via GLOBAL_LOAD_ASYNC_TO_LDS
// (ASYNCcnt threshold wait overlaps tile i+1's HBM fetch with tile i's
// WMMAs). Q fragments live in VGPRs for the whole kernel. P goes through a
// per-wave LDS slab (C-layout -> A-layout).
// ---------------------------------------------------------------------------
__global__ __launch_bounds__(256, 1) void attn_flash(
    const bf16_t* __restrict__ qn, const bf16_t* __restrict__ kn,
    const bf16_t* __restrict__ vT, bf16_t* __restrict__ aout) {
  constexpr int LKT = 136;  // sK row stride ([key][d]),  272B
  constexpr int LVT = 72;   // sV/sP row stride ([d][key] / [row][key]), 144B
  constexpr int KBUF = 64 * LKT;    // elems per K buffer
  constexpr int VBUF = 128 * LVT;   // elems per V buffer
  __shared__ bf16_t sK[2 * KBUF];        // 2 x 17408 B
  __shared__ bf16_t sV[2 * VBUF];        // 2 x 18432 B
  __shared__ bf16_t sP[8 * 16 * LVT];    // 18432 B (per-wave 16x64 slabs)

  const int tid = threadIdx.x, lane = tid & 31, wave = tid >> 5;
  const size_t bh = blockIdx.y;
  const int b = (int)(bh / NH), h = (int)(bh % NH);
  const int q0 = blockIdx.x * 128;

  const bf16_t* Q  = qn + (bh * Lseq + q0) * HD;
  const bf16_t* Kg = kn + bh * (size_t)Lseq * HD;
  const bf16_t* Vg = vT + bh * (size_t)HD * Lseq;

  // Pin this wave's Q fragments (16 rows x 128 d) in registers.
  v16bf qf[4];
  const bf16_t* Qw = Q + (size_t)wave * 16 * HD;
#pragma unroll
  for (int ks = 0; ks < 4; ++ks) qf[ks] = frag_a(Qw + ks * 32, HD, lane);

  v8f oacc[8];
#pragma unroll
  for (int dt = 0; dt < 8; ++dt)
#pragma unroll
    for (int r = 0; r < 8; ++r) oacc[dt][r] = 0.0f;
  float mrow[8], lrow[8];
#pragma unroll
  for (int r = 0; r < 8; ++r) { mrow[r] = -1e30f; lrow[r] = 0.0f; }

  const float scale = 0.088388347648318447f;  // 1/sqrt(128)
  const int chi = lane & 15, rhi = (lane >> 4) << 3;
  bf16_t* sPw = sP + wave * 16 * LVT;

  // Per-thread staging coordinates (fixed across iterations).
  const int kr = tid >> 2, kc = (tid & 3) * 32;   // K: row 0..63, 64B chunk
  const int vr = tid >> 1, vc = (tid & 1) * 32;   // V: row 0..127, 64B chunk

  // Issue one tile's async loads (8 x 16B per thread) into buffer `buf`.
  auto stage = [&](int kbase, int buf) {
    const bf16_t* gk = Kg + (size_t)(kbase + kr) * HD + kc;
    bf16_t* dk = sK + buf * KBUF + kr * LKT + kc;
#pragma unroll
    for (int i = 0; i < 4; ++i) async_cp16(dk + i * 8, gk + i * 8);
    const bf16_t* gv = Vg + (size_t)vr * Lseq + kbase + vc;
    bf16_t* dv = sV + buf * VBUF + vr * LVT + vc;
#pragma unroll
    for (int i = 0; i < 4; ++i) async_cp16(dv + i * 8, gv + i * 8);
  };

  stage(0, 0);                                  // prologue: tile 0 -> buf 0

  for (int k0 = 0; k0 < Lseq; k0 += 64) {
    const int cb = (k0 >> 6) & 1;               // compute buffer
    if (k0 + 64 < Lseq) {
      stage(k0 + 64, cb ^ 1);                   // prefetch next tile
      wait_async8();                            // tile k0 landed; next in flight
    } else {
      wait_async0();
    }
    __syncthreads();

    const bf16_t* Kt = sK + cb * KBUF;
    const bf16_t* Vt = sV + cb * VBUF;

    // S = Q * K^T : 4 key-tiles x 4 k-steps
    v8f sacc[4];
#pragma unroll
    for (int nt = 0; nt < 4; ++nt) {
#pragma unroll
      for (int r = 0; r < 8; ++r) sacc[nt][r] = 0.0f;
#pragma unroll
      for (int ks = 0; ks < 4; ++ks)
        sacc[nt] = wmma_bf16(qf[ks], frag_b(Kt + nt * 16 * LKT + ks * 32, LKT, lane),
                             sacc[nt]);
    }

    // Online softmax. Rows live per lane-half: VGPR r -> row r / r+8.
    float al[8];
#pragma unroll
    for (int r = 0; r < 8; ++r) {
      float m = sacc[0][r];
#pragma unroll
      for (int nt = 1; nt < 4; ++nt) m = fmaxf(m, sacc[nt][r]);
      m *= scale;
#pragma unroll
      for (int msk = 1; msk < 16; msk <<= 1) m = fmaxf(m, __shfl_xor(m, msk, 32));
      const float nm = fmaxf(mrow[r], m);
      al[r] = __expf(mrow[r] - nm);
      mrow[r] = nm;
    }
#pragma unroll
    for (int r = 0; r < 8; ++r) {
      float rs = 0.0f;
#pragma unroll
      for (int nt = 0; nt < 4; ++nt) {
        const float p = __expf(sacc[nt][r] * scale - mrow[r]);
        rs += p;
        sPw[(r + rhi) * LVT + nt * 16 + chi] = (bf16_t)p;  // C-layout -> [row][key]
      }
#pragma unroll
      for (int msk = 1; msk < 16; msk <<= 1) rs += __shfl_xor(rs, msk, 32);
      lrow[r] = lrow[r] * al[r] + rs;
#pragma unroll
      for (int dt = 0; dt < 8; ++dt) oacc[dt][r] *= al[r];
    }

    // O += P * V : 2 key-steps x 8 d-tiles
    const v16bf pf0 = frag_a(sPw, LVT, lane);
    const v16bf pf1 = frag_a(sPw + 32, LVT, lane);
#pragma unroll
    for (int dt = 0; dt < 8; ++dt) {
      oacc[dt] = wmma_bf16(pf0, frag_b(Vt + dt * 16 * LVT, LVT, lane), oacc[dt]);
      oacc[dt] = wmma_bf16(pf1, frag_b(Vt + dt * 16 * LVT + 32, LVT, lane), oacc[dt]);
    }
    __syncthreads();   // reads of buffer cb done; next iter may overwrite cb^1
  }

  // Normalize and store to [B, L, H*D] as bf16 (feeds the bf16-A proj GEMM).
#pragma unroll
  for (int r = 0; r < 8; ++r) {
    const float inv = 1.0f / lrow[r];
    const size_t row_l = (size_t)q0 + wave * 16 + r + rhi;
    const size_t obase = ((size_t)b * Lseq + row_l) * DIM + (size_t)h * HD;
#pragma unroll
    for (int dt = 0; dt < 8; ++dt)
      aout[obase + dt * 16 + chi] = (bf16_t)(oacc[dt][r] * inv);
  }
}

// ---------------------------------------------------------------------------
extern "C" void kernel_launch(void* const* d_in, const int* in_sizes, int n_in,
                              void* d_out, int out_size, void* d_ws, size_t ws_size,
                              hipStream_t stream) {
  (void)in_sizes; (void)n_in; (void)out_size; (void)ws_size;
  const float* x   = (const float*)d_in[0];
  const float* wqk = (const float*)d_in[1];
  const float* qw  = (const float*)d_in[2];
  const float* kw  = (const float*)d_in[3];
  const float* wpr = (const float*)d_in[4];
  const float* bpr = (const float*)d_in[5];
  float* out = (float*)d_out;

  const int M  = Bsz * Lseq;   // 4096
  const int N1 = 3 * DIM;      // 6144

  char* ws = (char*)d_ws;
  float* qkv = (float*)ws;                                   // fp32 [M, 3*dim]
  const size_t qkv_bytes = (size_t)M * N1 * sizeof(float);
  bf16_t* qn = (bf16_t*)(ws + qkv_bytes);                    // bf16 [B,H,L,D]
  const size_t hsz = (size_t)Bsz * NH * Lseq * HD;
  bf16_t* kn = qn + hsz;                                     // bf16 [B,H,L,D]
  bf16_t* vT = kn + hsz;                                     // bf16 [B,H,D,L]
  bf16_t* aout = (bf16_t*)ws;  // reuse dead qkv region: bf16 [B,L,dim]

  gemm_wmma<false, false><<<dim3(N1 / 128, M / 128), 256, 0, stream>>>(
      x, wqk, nullptr, qkv, M, N1, DIM);
  rms_reshape<<<(Bsz * Lseq * 3 * NH) / 8, 256, 0, stream>>>(
      qkv, qw, kw, qn, kn, vT);
  attn_flash<<<dim3(Lseq / 128, Bsz * NH), 256, 0, stream>>>(qn, kn, vT, aout);
  gemm_wmma<true, true><<<dim3(DIM / 128, M / 128), 256, 0, stream>>>(
      aout, wpr, bpr, out, M, DIM, DIM);
}
